// ContrastLoss_59871844106559
// MI455X (gfx1250) — matrix operand order repaired and checked by reading, not compiled, not run on hardware
//
#include <hip/hip_runtime.h>

typedef __attribute__((ext_vector_type(16))) _Float16 v16h;
typedef __attribute__((ext_vector_type(8)))  _Float16 v8h;
typedef __attribute__((ext_vector_type(8)))  float    v8f;

#define DT          1024                      // window length
#define FB          80                        // band bins (freq idx 23..102)
#define F0          23
#define NCOMP       160                       // 80 cos + 80 sin columns
#define NTILES      10                        // NCOMP / 16
#define KSTEPS      32                        // DT / 32 (K per WMMA)
#define TWID_HALVES (NTILES * KSTEPS * 32 * 16)   // 163840 f16
#define TWID_BYTES  (TWID_HALVES * 2)             // 327680 B (1280*256 -> keeps acc aligned)
#define NACC        162                       // sq[2] + s[2][80]

// ---------------------------------------------------------------------------
// Kernel 1: build DFT twiddles directly in WMMA B-fragment order + zero accs.
// Dense f16 B (32x16) layout assumed per ISA 7.12.2: element h of the v16h
// operand holds (K = (h&7) + 16*(h>=8) + 8*(lane>=16), N = lane%16).
// Column n<80 -> cos(2*pi*f*k/1024), n>=80 -> sin(...), f = 23 + (n mod 80).
// (Sign of the imaginary part is irrelevant: PSD squares it.)
// ---------------------------------------------------------------------------
__global__ void twid_init_kernel(_Float16* __restrict__ twid,
                                 float* __restrict__ acc) {
  int gid = blockIdx.x * blockDim.x + threadIdx.x;
  if (gid < NACC) acc[gid] = 0.0f;
  for (int i = gid; i < TWID_HALVES; i += gridDim.x * blockDim.x) {
    int h    = i & 15;
    int lane = (i >> 4) & 31;
    int s    = (i >> 9) & 31;
    int nt   = i >> 14;
    int K    = s * 32 + (h & 7) + ((h & 8) ? 16 : 0) + ((lane & 16) ? 8 : 0);
    int n    = nt * 16 + (lane & 15);
    int f    = F0 + ((n < FB) ? n : n - FB);
    int ph   = (f * K) & (DT - 1);            // exact integer phase reduction
    float th = (float)ph * (6.283185307179586f / (float)DT);
    float v  = (n < FB) ? __cosf(th) : __sinf(th);
    twid[i] = (_Float16)v;
  }
}

// ---------------------------------------------------------------------------
// Kernel 2: per (b,c): gather 16 windows -> LDS f16, band-DFT via WMMA,
// PSD + normalize, accumulate sq[b] and s[b][f].
// ---------------------------------------------------------------------------
__global__ __launch_bounds__(320)
void dft_psd_kernel(const float* __restrict__ x,     // [2,256,8192]
                    const int*   __restrict__ offs,  // [2,256,16] flat == seg id
                    const _Float16* __restrict__ twid,
                    float* __restrict__ acc) {       // [2 + 2*80]
  __shared__ __align__(32) _Float16 As[16][DT + 8];  // +8 halves row pad
  __shared__ float sqs[16][NCOMP];
  __shared__ float band[16][FB];
  __shared__ float rowinv[16];
  __shared__ int   rowoff[16];
  __shared__ float blk_sq;

  const int tid = threadIdx.x;
  const int bc  = blockIdx.x;        // b*256 + c  (16 segs per block)
  const int b   = bc >> 8;

  if (tid < 16) rowoff[tid] = offs[(bc << 4) + tid];
  if (tid == 0) blk_sq = 0.0f;
  __syncthreads();

  // Stage 16 windows (f32 -> f16). Mean subtraction provably unnecessary.
  const float* xrow = x + (size_t)bc * 8192;
  for (int i = tid; i < 16 * DT; i += 320) {
    int r = i >> 10, c = i & (DT - 1);
    As[r][c] = (_Float16)xrow[rowoff[r] + c];
  }
  __syncthreads();

  // ---- WMMA main loop: wave w owns N-tile w (16 of the 160 components) ----
  const int wave = tid >> 5;               // 0..9
  const int lane = tid & 31;
  const _Float16* Arow = &As[lane & 15][0];
  const int koff = (lane & 16) ? 8 : 0;    // A-fragment lane split (ISA 7.12.2)
  const _Float16* Bp = twid + ((size_t)(wave * KSTEPS) * 32 + lane) * 16;

  v8f cacc = {};
  #pragma unroll 4
  for (int s = 0; s < KSTEPS; ++s) {
    const int kb = s * 32 + koff;
    v8h lo = *(const v8h*)(Arow + kb);        // K = kb .. kb+7
    v8h hi = *(const v8h*)(Arow + kb + 16);   // K = kb+16 .. kb+23
    v16h a;
    #pragma unroll
    for (int i = 0; i < 8; ++i) { a[i] = lo[i]; a[8 + i] = hi[i]; }
    v16h bm = *(const v16h*)(Bp + (size_t)s * (32 * 16));
    cacc = __builtin_amdgcn_wmma_f32_16x16x32_f16(
        false, a, false, bm, (short)0, cacc, false, false);
  }

  // ---- epilogue: PSD, band combine, normalize, reduce ----
  {
    const int nb = wave * 16 + (lane & 15);
    const int mb = (lane & 16) ? 8 : 0;      // C layout: VGPR v -> M = v (+8)
    #pragma unroll
    for (int v = 0; v < 8; ++v) sqs[mb + v][nb] = cacc[v] * cacc[v];
  }
  __syncthreads();

  for (int i = tid; i < 16 * FB; i += 320) {
    int r = i / FB, f = i - r * FB;
    band[r][f] = sqs[r][f] + sqs[r][FB + f]; // Re^2 + Im^2
  }
  __syncthreads();

  if (tid < 16) {
    float ssum = 0.0f;
    #pragma unroll 8
    for (int f = 0; f < FB; ++f) ssum += band[tid][f];
    rowinv[tid] = 1.0f / ssum;
  }
  __syncthreads();

  if (tid < FB) {
    float colsum = 0.0f, sqp = 0.0f;
    #pragma unroll 4
    for (int r = 0; r < 16; ++r) {
      float p = band[r][tid] * rowinv[r];
      colsum += p;
      sqp    += p * p;
    }
    atomicAdd(&acc[2 + b * FB + tid], colsum);  // s[b][f]
    atomicAdd(&blk_sq, sqp);                    // LDS ds_add_f32
  }
  __syncthreads();
  if (tid == 0) atomicAdd(&acc[b], blk_sq);     // sq[b]
}

// ---------------------------------------------------------------------------
// Kernel 3: closed-form losses from sq[2] and s[2][80].
// ---------------------------------------------------------------------------
__global__ void finalize_kernel(const float* __restrict__ acc,
                                float* __restrict__ out) {
  if (threadIdx.x == 0) {
    const float N = 4096.0f, Fb = (float)FB;
    float sq0 = acc[0], sq1 = acc[1];
    float ss0 = 0.0f, ss1 = 0.0f, s01 = 0.0f;
    for (int f = 0; f < FB; ++f) {
      float a0 = acc[2 + f], a1 = acc[2 + FB + f];
      ss0 += a0 * a0; ss1 += a1 * a1; s01 += a0 * a1;
    }
    float pos0 = (2.0f * N * sq0 - 2.0f * ss0) / Fb / (N * N - N);
    float pos1 = (2.0f * N * sq1 - 2.0f * ss1) / Fb / (N * N - N);
    float pos  = 0.5f * (pos0 + pos1);
    float neg  = -(N * sq0 + N * sq1 - 2.0f * s01) / Fb / (N * N);
    out[0] = pos + neg;   // loss
    out[1] = pos;         // pos_loss
    out[2] = neg;         // neg_loss
  }
}

extern "C" void kernel_launch(void* const* d_in, const int* in_sizes, int n_in,
                              void* d_out, int out_size, void* d_ws, size_t ws_size,
                              hipStream_t stream) {
  (void)in_sizes; (void)n_in; (void)out_size; (void)ws_size;
  const float* x    = (const float*)d_in[0];
  const int*   offs = (const int*)d_in[1];
  float*       out  = (float*)d_out;
  _Float16*    twid = (_Float16*)d_ws;
  float*       acc  = (float*)((char*)d_ws + TWID_BYTES);

  twid_init_kernel<<<640, 256, 0, stream>>>(twid, acc);
  dft_psd_kernel<<<512, 320, 0, stream>>>(x, offs, twid, acc);
  finalize_kernel<<<1, 32, 0, stream>>>(acc, out);
}
